// Head_50594714747030
// MI455X (gfx1250) — matrix operand order repaired and checked by reading, not compiled, not run on hardware
//
#include <hip/hip_runtime.h>
#include <hip/hip_bf16.h>

typedef __attribute__((ext_vector_type(16))) __bf16 v16bf;
typedef __attribute__((ext_vector_type(8)))  __bf16 v8bf;
typedef __attribute__((ext_vector_type(8)))  float  v8f;
typedef __attribute__((ext_vector_type(4)))  float  v4f;

#define WMMA_BF16(a, b, c) \
  __builtin_amdgcn_wmma_f32_16x16x32_bf16(false, (a), false, (b), (short)0, (c), false, false)

static constexpr int Tn = 2048;
static constexpr int En = 1024;
static constexpr int Hn = 64;
static constexpr int Bn = 8;

// workspace layout in bf16 elements
static constexpr size_t WT_OFF = 0;                               // 3*64*1024
static constexpr size_t K_OFF  = (size_t)3 * Hn * En;             // B*T*H (pre-scaled by 1/sqrt(H))
static constexpr size_t Q_OFF  = K_OFF + (size_t)Bn * Tn * Hn;
static constexpr size_t V_OFF  = Q_OFF + (size_t)Bn * Tn * Hn;    // stored transposed [B][H][T]

// ---- fragment loaders --------------------------------------------------
// B fragment (32x16 bf16): lane L holds column n=L%16; lanes 0-15 carry
// K=0..15, lanes 16-31 carry K=16..31 -> 16 contiguous bf16 per lane.
__device__ inline v16bf load_b_frag(const __bf16* base) {
  const v8bf* p = (const v8bf*)base;
  v8bf lo = p[0], hi = p[1];
  v16bf r;
#pragma unroll
  for (int i = 0; i < 8; ++i) { r[i] = lo[i]; r[8 + i] = hi[i]; }
  return r;
}

// A fragment (16x32 bf16): lane L holds row M=L%16; elements 0..15 are
// K = (L<16 ? 0 : 16)+e (contiguous, matching B fragments).
__device__ inline v16bf load_a_frag_bf16(const __bf16* row, int k0, int alo) {
  const v8bf* p0 = (const v8bf*)(row + k0 + alo);
  const v8bf* p1 = (const v8bf*)(row + k0 + 8 + alo);
  v8bf lo = p0[0], hi = p1[0];
  v16bf r;
#pragma unroll
  for (int i = 0; i < 8; ++i) { r[i] = lo[i]; r[8 + i] = hi[i]; }
  return r;
}

// Same A layout but sourced from fp32 global memory with on-the-fly bf16 cvt.
__device__ inline v16bf load_a_frag_f32(const float* row, int k0, int alo) {
  const v4f* p0 = (const v4f*)(row + k0 + alo);
  const v4f* p1 = (const v4f*)(row + k0 + 8 + alo);
  v4f x0 = p0[0], x1 = p0[1], x2 = p1[0], x3 = p1[1];
  v16bf r;
#pragma unroll
  for (int i = 0; i < 4; ++i) {
    r[i]      = (__bf16)x0[i];
    r[4 + i]  = (__bf16)x1[i];
    r[8 + i]  = (__bf16)x2[i];
    r[12 + i] = (__bf16)x3[i];
  }
  return r;
}

// ---- kernel 1: weight transpose + fp32->bf16 ---------------------------
// wT[m][n][k] = W_m[k][n]   (K-contiguous so B fragments load contiguously)
__global__ void wconv_kernel(const float* __restrict__ Wk,
                             const float* __restrict__ Wq,
                             const float* __restrict__ Wv,
                             __bf16* __restrict__ wT) {
  int idx = blockIdx.x * blockDim.x + threadIdx.x;
  if (idx >= 3 * Hn * En) return;
  int m   = idx >> 16;        // /65536
  int rem = idx & 65535;
  int n   = rem >> 10;        // /1024
  int kk  = rem & 1023;
  const float* W = (m == 0) ? Wk : ((m == 1) ? Wq : Wv);
  wT[idx] = (__bf16)W[kk * Hn + n];
}

// ---- kernel 2: fused k/q/v projection ----------------------------------
// One wave per 16-row tile of [B*T]; K-loop of 32 over E; 12 accumulators.
// k output is pre-scaled by 1/sqrt(H) so attention needs no scale.
__global__ void __launch_bounds__(128)
proj_kernel(const float* __restrict__ x, const __bf16* __restrict__ wT,
            __bf16* __restrict__ kout, __bf16* __restrict__ qout,
            __bf16* __restrict__ vT) {
  int lane = threadIdx.x & 31;
  int wave = threadIdx.x >> 5;
  int tile = blockIdx.x * 4 + wave;   // 0 .. B*T/16-1
  int r0 = tile * 16;
  int lr = lane & 15;
  int alo = (lane < 16) ? 0 : 16;
  int blo = (lane < 16) ? 0 : 16;
  int rhalf = (lane >> 4) * 8;

  const float* arow = x + (size_t)(r0 + lr) * En;

  v8f zero = {0.f, 0.f, 0.f, 0.f, 0.f, 0.f, 0.f, 0.f};
  v8f ak[4], aq[4], av[4];
#pragma unroll
  for (int nt = 0; nt < 4; ++nt) { ak[nt] = zero; aq[nt] = zero; av[nt] = zero; }

  for (int k0 = 0; k0 < En; k0 += 32) {
    v16bf a = load_a_frag_f32(arow, k0, alo);
#pragma unroll
    for (int nt = 0; nt < 4; ++nt) {
      int n = nt * 16 + lr;
      const __bf16* wk = wT + (size_t)n * En + k0 + blo;
      const __bf16* wq = wk + (size_t)Hn * En;        // m=1 block
      const __bf16* wv = wk + (size_t)2 * Hn * En;    // m=2 block
      ak[nt] = WMMA_BF16(a, load_b_frag(wk), ak[nt]);
      aq[nt] = WMMA_BF16(a, load_b_frag(wq), aq[nt]);
      av[nt] = WMMA_BF16(a, load_b_frag(wv), av[nt]);
    }
  }

  const float SC = 0.125f;    // 1/sqrt(64) folded into k
  int b   = r0 / Tn;
  int t0l = r0 % Tn;
#pragma unroll
  for (int nt = 0; nt < 4; ++nt) {
    int n = nt * 16 + lr;
#pragma unroll
    for (int v = 0; v < 8; ++v) {
      int row = r0 + v + rhalf;           // D layout: M = v + 8*(lane>=16)
      kout[(size_t)row * Hn + n] = (__bf16)(ak[nt][v] * SC);
      qout[(size_t)row * Hn + n] = (__bf16)aq[nt][v];
      int t = t0l + v + rhalf;
      vT[((size_t)b * Hn + n) * Tn + t] = (__bf16)av[nt][v];
    }
  }
}

// ---- kernel 3: causal flash attention (wei = k . q^T) ------------------
// One wave per 16 t-rows; 64-wide s-chunks (4 S subtiles) to amortize the
// softmax shuffles/rescales across twice as many columns per iteration.
__global__ void __launch_bounds__(128)
attn_kernel(const __bf16* __restrict__ kb, const __bf16* __restrict__ qb,
            const __bf16* __restrict__ vT, float* __restrict__ out) {
  __shared__ __align__(16) __bf16 Plds[4][16][64];   // wave-private P tiles

  int lane = threadIdx.x & 31;
  int wave = threadIdx.x >> 5;
  int tile = blockIdx.x * 4 + wave;   // 0..1023
  int b  = tile >> 7;                 // / (T/16)
  int t0 = (tile & 127) * 16;
  int lr = lane & 15;
  int alo = (lane < 16) ? 0 : 16;
  int blo = (lane < 16) ? 0 : 16;
  int rhalf = (lane >> 4) * 8;

  // k rows act as "queries": load the two H-chunk A fragments once.
  const __bf16* krow = kb + ((size_t)b * Tn + t0 + lr) * Hn;
  v16bf aK0 = load_a_frag_bf16(krow, 0, alo);
  v16bf aK1 = load_a_frag_bf16(krow, 32, alo);

  v8f zero = {0.f, 0.f, 0.f, 0.f, 0.f, 0.f, 0.f, 0.f};
  v8f acc[4];
#pragma unroll
  for (int ht = 0; ht < 4; ++ht) acc[ht] = zero;
  float m[8], l[8];
#pragma unroll
  for (int v = 0; v < 8; ++v) { m[v] = -1e30f; l[v] = 0.f; }

  const float L2E = 1.4426950408889634f;
  const __bf16* vbase = vT + (size_t)b * Hn * Tn;
  const __bf16* qbat  = qb + (size_t)b * Tn * Hn;

  // s0 is 64-aligned and s0 <= t0+15 <= 2047 implies s0 <= 1984, so every
  // q-row / vT-column index below stays strictly within [0, 2047].
  for (int s0 = 0; s0 <= t0 + 15; s0 += 64) {
    // ---- S = k-tile @ q^T : 4 column subtiles x 2 H chunks
    v8f S[4];
#pragma unroll
    for (int j = 0; j < 4; ++j) {
      const __bf16* qr = qbat + (size_t)(s0 + 16 * j + lr) * Hn;
      S[j] = zero;
      S[j] = WMMA_BF16(aK0, load_b_frag(qr + blo), S[j]);
      S[j] = WMMA_BF16(aK1, load_b_frag(qr + 32 + blo), S[j]);
    }

    // ---- prefetch next chunk's streams (wave-uniform branch)
    if (s0 + 64 <= t0 + 15) {
      const __bf16* qn = qbat + (size_t)(s0 + 64 + lr) * Hn;
#pragma unroll
      for (int j = 0; j < 4; ++j)
        __builtin_prefetch((const void*)(qn + (size_t)(16 * j) * Hn), 0, 3);
      const __bf16* vn = vbase + (size_t)(2 * lane) * Tn + s0 + 64;
      __builtin_prefetch((const void*)vn, 0, 3);
      __builtin_prefetch((const void*)(vn + Tn), 0, 3);
    }

    // ---- online softmax (row M = v + rhalf lives in one 16-lane half)
    float p[4][8];
#pragma unroll
    for (int v = 0; v < 8; ++v) {
      int t = t0 + v + rhalf;
      float x0 = (s0 + lr      <= t) ? S[0][v] : -1e30f;
      float x1 = (s0 + 16 + lr <= t) ? S[1][v] : -1e30f;
      float x2 = (s0 + 32 + lr <= t) ? S[2][v] : -1e30f;
      float x3 = (s0 + 48 + lr <= t) ? S[3][v] : -1e30f;
      float rm = fmaxf(fmaxf(x0, x1), fmaxf(x2, x3));
      rm = fmaxf(rm, __shfl_xor(rm, 1, 32));
      rm = fmaxf(rm, __shfl_xor(rm, 2, 32));
      rm = fmaxf(rm, __shfl_xor(rm, 4, 32));
      rm = fmaxf(rm, __shfl_xor(rm, 8, 32));
      float mn  = fmaxf(m[v], rm);
      float esc = __builtin_amdgcn_exp2f((m[v] - mn) * L2E);
      float e0  = __builtin_amdgcn_exp2f((x0 - mn) * L2E);
      float e1  = __builtin_amdgcn_exp2f((x1 - mn) * L2E);
      float e2  = __builtin_amdgcn_exp2f((x2 - mn) * L2E);
      float e3  = __builtin_amdgcn_exp2f((x3 - mn) * L2E);
      float rs = (e0 + e1) + (e2 + e3);
      rs += __shfl_xor(rs, 1, 32);
      rs += __shfl_xor(rs, 2, 32);
      rs += __shfl_xor(rs, 4, 32);
      rs += __shfl_xor(rs, 8, 32);
      l[v] = l[v] * esc + rs;
      m[v] = mn;
      acc[0][v] *= esc; acc[1][v] *= esc; acc[2][v] *= esc; acc[3][v] *= esc;
      p[0][v] = e0; p[1][v] = e1; p[2][v] = e2; p[3][v] = e3;
    }

    // ---- D-layout -> A-layout via wave-private LDS
#pragma unroll
    for (int j = 0; j < 4; ++j)
#pragma unroll
      for (int v = 0; v < 8; ++v)
        Plds[wave][v + rhalf][16 * j + lr] = (__bf16)p[j][v];
    asm volatile("s_wait_dscnt 0" ::: "memory");   // same-wave DS ordering
    v16bf aP0 = load_a_frag_bf16(&Plds[wave][lr][0], 0, alo);
    v16bf aP1 = load_a_frag_bf16(&Plds[wave][lr][0], 32, alo);

    // ---- acc += P @ V   (V stored transposed: rows are h, cols are s)
#pragma unroll
    for (int ht = 0; ht < 4; ++ht) {
      const __bf16* vp = vbase + (size_t)(ht * 16 + lr) * Tn + s0;
      acc[ht] = WMMA_BF16(aP0, load_b_frag(vp + blo), acc[ht]);
      acc[ht] = WMMA_BF16(aP1, load_b_frag(vp + 32 + blo), acc[ht]);
    }
  }

  // ---- epilogue: out[b][t][h] = acc / l (one rcp per row)
  float inv[8];
#pragma unroll
  for (int v = 0; v < 8; ++v) inv[v] = __builtin_amdgcn_rcpf(l[v]);
#pragma unroll
  for (int ht = 0; ht < 4; ++ht) {
#pragma unroll
    for (int v = 0; v < 8; ++v) {
      int t = t0 + v + rhalf;
      out[((size_t)b * Tn + t) * Hn + ht * 16 + lr] = acc[ht][v] * inv[v];
    }
  }
}

// ---- launch ------------------------------------------------------------
extern "C" void kernel_launch(void* const* d_in, const int* in_sizes, int n_in,
                              void* d_out, int out_size, void* d_ws, size_t ws_size,
                              hipStream_t stream) {
  const float* x  = (const float*)d_in[0];
  const float* Wk = (const float*)d_in[1];
  const float* Wq = (const float*)d_in[2];
  const float* Wv = (const float*)d_in[3];
  __bf16* ws = (__bf16*)d_ws;
  __bf16* wT = ws + WT_OFF;
  __bf16* kb = ws + K_OFF;
  __bf16* qb = ws + Q_OFF;
  __bf16* vt = ws + V_OFF;
  float* out = (float*)d_out;

  wconv_kernel<<<(3 * Hn * En + 255) / 256, 256, 0, stream>>>(Wk, Wq, Wv, wT);
  // B*T/16 = 1024 tiles, 4 waves (128 threads) per block
  proj_kernel<<<256, 128, 0, stream>>>(x, wT, kb, qb, vt);
  attn_kernel<<<256, 128, 0, stream>>>(kb, qb, vt, out);
}
